// MultiHead_20495583936536
// MI455X (gfx1250) — matrix operand order repaired and checked by reading, not compiled, not run on hardware
//
#include <hip/hip_runtime.h>

// ---------------------------------------------------------------------------
// MultiHead attention, MI455X (gfx1250), bf16 WMMA pipeline.
// B=2, S=2048, D=1024, H=16, HD=64. All matmuls via v_wmma_f32_16x16x32_bf16.
// K-block staging uses the CDNA5 async global->LDS path (ASYNCcnt) via inline
// asm; mask stream prefetched via global_prefetch.
// ---------------------------------------------------------------------------

typedef __bf16 bf16;
typedef __attribute__((ext_vector_type(8)))  __bf16 v8bf;
typedef __attribute__((ext_vector_type(16))) __bf16 v16bf;
typedef __attribute__((ext_vector_type(8)))  float  v8f;
typedef __attribute__((ext_vector_type(4)))  float  f32x4;

#if defined(__gfx1250__)
#define HAVE_ASYNC_LDS 1
#else
#define HAVE_ASYNC_LDS 0
#endif

__device__ __forceinline__ void async_copy16_to_lds(const void* gsrc, void* lds_dst) {
#if HAVE_ASYNC_LDS
  // GV mode: 64-bit global address in VGPR pair, LDS byte address in VDST.
  // Low 32 bits of the flat shared-memory pointer == LDS offset (aperture).
  unsigned lds_addr = (unsigned)(unsigned long long)lds_dst;
  asm volatile("global_load_async_to_lds_b128 %0, %1, off"
               :: "v"(lds_addr), "v"(gsrc)
               : "memory");
#else
  *(v8bf*)lds_dst = *(const v8bf*)gsrc;
#endif
}

__device__ __forceinline__ void wait_async_zero() {
#if HAVE_ASYNC_LDS
#if __has_builtin(__builtin_amdgcn_s_wait_asynccnt)
  __builtin_amdgcn_s_wait_asynccnt(0);
#else
  asm volatile("s_wait_asynccnt 0x0" ::: "memory");
#endif
#endif
}

#define WMMA_BF16(A, B, C) \
  __builtin_amdgcn_wmma_f32_16x16x32_bf16(false, (A), false, (B), (short)0, (C), false, false)

__device__ __forceinline__ v16bf make_v16(const v8bf lo, const v8bf hi) {
  v16bf r;
#pragma unroll
  for (int e = 0; e < 8; ++e) { r[e] = lo[e]; r[8 + e] = hi[e]; }
  return r;
}

__device__ __forceinline__ float hmax16(float v) {
#pragma unroll
  for (int off = 1; off < 16; off <<= 1) v = fmaxf(v, __shfl_xor(v, off, 32));
  return v;
}
__device__ __forceinline__ float hsum16(float v) {
#pragma unroll
  for (int off = 1; off < 16; off <<= 1) v += __shfl_xor(v, off, 32);
  return v;
}

// ---------------------------------------------------------------------------
// GEMM: out[M,N] = A[M,K] @ W[K,N] + bias[N].  A,W f32 in memory, converted
// to bf16 on the fly; accumulate f32 via WMMA.  64x64 block tile, 128 threads
// (4 waves), each wave owns a 16x64 strip (1 A-frag + 4 WMMAs per K-step).
// ---------------------------------------------------------------------------
template <typename OutT>
__global__ __launch_bounds__(128) void gemm_bias_kernel(
    const float* __restrict__ A, const float* __restrict__ W,
    const float* __restrict__ bias, OutT* __restrict__ out,
    int M, int N, int K) {
  __shared__ bf16 As[64][40];  // [m][k], stride 40 (80B) keeps 16B alignment
  __shared__ bf16 Bs[64][40];  // [n][k]  (W transposed in LDS)

  const int t    = threadIdx.x;
  const int lane = t & 31;
  const int wave = t >> 5;
  const int half = lane >> 4;   // K-half selector per WMMA fragment layout
  const int l16  = lane & 15;
  const int m0   = blockIdx.x * 64;
  const int n0   = blockIdx.y * 64;

  v8f acc[4];
#pragma unroll
  for (int nt = 0; nt < 4; ++nt)
#pragma unroll
    for (int j = 0; j < 8; ++j) acc[nt][j] = 0.0f;

  for (int k0 = 0; k0 < K; k0 += 32) {
    // Stage A tile 64x32 (f32 -> bf16), 4 float4 per thread.
#pragma unroll
    for (int i = 0; i < 4; ++i) {
      int c   = t + 128 * i;        // 0..511 float4 chunks
      int row = c >> 3;
      int c4  = (c & 7) * 4;
      f32x4 v = *(const f32x4*)(A + (size_t)(m0 + row) * K + k0 + c4);
#pragma unroll
      for (int e = 0; e < 4; ++e) As[row][c4 + e] = (bf16)v[e];
      if (i == 0 && k0 + 32 < K)    // hint next A tile into cache
        __builtin_prefetch(A + (size_t)(m0 + row) * K + k0 + 32 + c4, 0, 0);
    }
    // Stage W tile 32x64 transposed into Bs[n][k].
#pragma unroll
    for (int i = 0; i < 4; ++i) {
      int c  = t + 128 * i;
      int kr = c >> 4;              // 0..31
      int n4 = (c & 15) * 4;
      f32x4 v = *(const f32x4*)(W + (size_t)(k0 + kr) * N + n0 + n4);
#pragma unroll
      for (int e = 0; e < 4; ++e) Bs[n4 + e][kr] = (bf16)v[e];
    }
    __syncthreads();

    // A fragment: 16x32, lane l -> row l16; elems 0-7: K=(half?8:0)+e,
    // elems 8-15: K=16+(half?8:0)+e  -> two 16B LDS loads.
    const bf16* ar = &As[wave * 16 + l16][0];
    v16bf afrag = make_v16(*(const v8bf*)(ar + (half ? 8 : 0)),
                           *(const v8bf*)(ar + 16 + (half ? 8 : 0)));
#pragma unroll
    for (int nt = 0; nt < 4; ++nt) {
      // B fragment: 32x16, lane l -> col l16; elems 0-7: K=(half?8:0)+e,
      // elems 8-15: K=16+(half?8:0)+e.
      const bf16* br = &Bs[nt * 16 + l16][0];
      v16bf bfrag = make_v16(*(const v8bf*)(br + (half ? 8 : 0)),
                             *(const v8bf*)(br + 16 + (half ? 8 : 0)));
      acc[nt] = WMMA_BF16(afrag, bfrag, acc[nt]);
    }
    __syncthreads();
  }

  // Epilogue: C layout lane l -> col l16, VGPR j -> row j + 8*half.
#pragma unroll
  for (int nt = 0; nt < 4; ++nt) {
    int n = n0 + nt * 16 + l16;
    float bv = bias[n];
#pragma unroll
    for (int j = 0; j < 8; ++j) {
      int m = m0 + wave * 16 + j + 8 * half;
      out[(size_t)m * N + n] = (OutT)(acc[nt][j] + bv);
    }
  }
}

// ---------------------------------------------------------------------------
// Flash attention: grid (S/64, B*H), 128 threads (4 waves).  Each wave owns
// 16 query rows; Q fragments stay in registers across the whole KV sweep.
// K staged to LDS via async copies (overlapped with the V transpose work),
// online softmax, P restaged via per-wave LDS buffer.
// ---------------------------------------------------------------------------
__global__ __launch_bounds__(128) void attn_kernel(
    const bf16* __restrict__ Q, const bf16* __restrict__ Kb,
    const bf16* __restrict__ Vb, const float* __restrict__ mask,
    float* __restrict__ ctx) {
  constexpr int S = 2048, D = 1024, H = 16, HD = 64;
  __shared__ bf16 Ks[64][72];       // [kv][d]
  __shared__ bf16 Vt[64][72];       // [d][kv]  (transposed at stage time)
  __shared__ bf16 Ps[4][16][72];    // per-wave P staging [wave][q][kv]

  const int t    = threadIdx.x;
  const int lane = t & 31;
  const int wave = t >> 5;
  const int half = lane >> 4;
  const int l16  = lane & 15;
  const int b    = blockIdx.y / H;
  const int h    = blockIdx.y % H;
  const int q0   = blockIdx.x * 64 + wave * 16;  // wave's query base
  const float inv_scale = 1.0f / (sqrtf((float)HD) + 1e-8f);

  // Q fragments (16x64 split into two 16x32 chunks), loaded once.
  const bf16* qrow = Q + ((size_t)(b * S) + q0 + l16) * D + h * HD;
  v16bf aq0 = make_v16(*(const v8bf*)(qrow + (half ? 8 : 0)),
                       *(const v8bf*)(qrow + 16 + (half ? 8 : 0)));
  v16bf aq1 = make_v16(*(const v8bf*)(qrow + 32 + (half ? 8 : 0)),
                       *(const v8bf*)(qrow + 48 + (half ? 8 : 0)));

  float row_max[8], row_sum[8];
  v8f cacc[4];
#pragma unroll
  for (int j = 0; j < 8; ++j) { row_max[j] = -1e30f; row_sum[j] = 0.0f; }
#pragma unroll
  for (int dt = 0; dt < 4; ++dt)
#pragma unroll
    for (int j = 0; j < 8; ++j) cacc[dt][j] = 0.0f;

  for (int kv0 = 0; kv0 < S; kv0 += 64) {
    // ---- Stage K asynchronously (pure 16B copy -> GLOBAL_LOAD_ASYNC_TO_LDS),
    //      then do the V transpose on the VALU path while K is in flight.
#pragma unroll
    for (int i = 0; i < 4; ++i) {
      int c   = t + 128 * i;        // 0..511 v8bf chunks
      int row = c >> 3;
      int dc  = (c & 7) * 8;
      size_t g = ((size_t)(b * S) + kv0 + row) * D + h * HD + dc;
      async_copy16_to_lds(Kb + g, &Ks[row][dc]);
    }
#pragma unroll
    for (int i = 0; i < 4; ++i) {
      int c   = t + 128 * i;
      int row = c >> 3;
      int dc  = (c & 7) * 8;
      size_t g = ((size_t)(b * S) + kv0 + row) * D + h * HD + dc;
      v8bf vv = *(const v8bf*)(Vb + g);
#pragma unroll
      for (int e = 0; e < 8; ++e) Vt[dc + e][row] = vv[e];
    }
    // Prefetch the next KV block's mask rows (largest read stream).
    if (kv0 + 64 < S)
      __builtin_prefetch(mask + ((size_t)b * S + q0 + l16) * S + kv0 + 64, 0, 0);
    wait_async_zero();
    __syncthreads();

    // Scores: S16x64 = Q(16x64) . K^T, four 16x16 tiles.
    float p[4][8];
#pragma unroll
    for (int nt = 0; nt < 4; ++nt) {
      const bf16* kr = &Ks[nt * 16 + l16][0];  // row = kv, contiguous d
      v16bf bk0 = make_v16(*(const v8bf*)(kr + (half ? 8 : 0)),
                           *(const v8bf*)(kr + 16 + (half ? 8 : 0)));
      v16bf bk1 = make_v16(*(const v8bf*)(kr + 32 + (half ? 8 : 0)),
                           *(const v8bf*)(kr + 48 + (half ? 8 : 0)));
      v8f s;
#pragma unroll
      for (int j = 0; j < 8; ++j) s[j] = 0.0f;
      s = WMMA_BF16(aq0, bk0, s);
      s = WMMA_BF16(aq1, bk1, s);
#pragma unroll
      for (int j = 0; j < 8; ++j) {
        int qg = q0 + j + 8 * half;
        int kg = kv0 + nt * 16 + l16;
        p[nt][j] = s[j] * inv_scale +
                   mask[((size_t)b * S + qg) * S + kg] * (-1e9f);
      }
    }

    // Online softmax (rows live across 16 lanes of a half-wave).
    float alpha[8], psum[8];
#pragma unroll
    for (int j = 0; j < 8; ++j) {
      float m = fmaxf(fmaxf(p[0][j], p[1][j]), fmaxf(p[2][j], p[3][j]));
      m = hmax16(m);
      float nm = fmaxf(row_max[j], m);
      alpha[j] = __expf(row_max[j] - nm);
      row_max[j] = nm;
    }
#pragma unroll
    for (int j = 0; j < 8; ++j) {
      float ps = 0.0f;
#pragma unroll
      for (int nt = 0; nt < 4; ++nt) {
        p[nt][j] = __expf(p[nt][j] - row_max[j]);
        ps += p[nt][j];
      }
      psum[j] = hsum16(ps);
      row_sum[j] = row_sum[j] * alpha[j] + psum[j];
    }
#pragma unroll
    for (int dt = 0; dt < 4; ++dt)
#pragma unroll
      for (int j = 0; j < 8; ++j) cacc[dt][j] *= alpha[j];

    // Re-layout P (C-layout f32) -> A-fragment bf16 via per-wave LDS.
#pragma unroll
    for (int nt = 0; nt < 4; ++nt)
#pragma unroll
      for (int j = 0; j < 8; ++j)
        Ps[wave][j + 8 * half][nt * 16 + l16] = (bf16)p[nt][j];

    const bf16* pr = &Ps[wave][l16][0];
    v16bf ap0 = make_v16(*(const v8bf*)(pr + (half ? 8 : 0)),
                         *(const v8bf*)(pr + 16 + (half ? 8 : 0)));
    v16bf ap1 = make_v16(*(const v8bf*)(pr + 32 + (half ? 8 : 0)),
                         *(const v8bf*)(pr + 48 + (half ? 8 : 0)));

    // ctx += P(16x64) . V(64x64): B-frags from transposed V, contiguous kv.
#pragma unroll
    for (int dt = 0; dt < 4; ++dt) {
      const bf16* vr = &Vt[dt * 16 + l16][0];  // row = d, contiguous kv
      v16bf bv0 = make_v16(*(const v8bf*)(vr + (half ? 8 : 0)),
                           *(const v8bf*)(vr + 16 + (half ? 8 : 0)));
      v16bf bv1 = make_v16(*(const v8bf*)(vr + 32 + (half ? 8 : 0)),
                           *(const v8bf*)(vr + 48 + (half ? 8 : 0)));
      cacc[dt] = WMMA_BF16(ap0, bv0, cacc[dt]);
      cacc[dt] = WMMA_BF16(ap1, bv1, cacc[dt]);
    }
    __syncthreads();
  }

  // Normalize and write ctx back to [B,S,D] (f32) for the output projection.
#pragma unroll
  for (int dt = 0; dt < 4; ++dt) {
    int d = dt * 16 + l16;
#pragma unroll
    for (int j = 0; j < 8; ++j) {
      int qg = q0 + j + 8 * half;
      ctx[((size_t)(b * S) + qg) * D + h * HD + d] = cacc[dt][j] / row_sum[j];
    }
  }
}

// ---------------------------------------------------------------------------
// Launch: 3x QKV projection GEMMs -> flash attention -> output projection.
// ---------------------------------------------------------------------------
extern "C" void kernel_launch(void* const* d_in, const int* in_sizes, int n_in,
                              void* d_out, int out_size, void* d_ws, size_t ws_size,
                              hipStream_t stream) {
  constexpr int Bn = 2, S = 2048, D = 1024;
  constexpr int M = Bn * S;  // 4096

  const float* q    = (const float*)d_in[0];
  const float* k    = (const float*)d_in[1];
  const float* v    = (const float*)d_in[2];
  const float* mask = (const float*)d_in[3];
  const float* wq_w = (const float*)d_in[4];
  const float* wq_b = (const float*)d_in[5];
  const float* wk_w = (const float*)d_in[6];
  const float* wk_b = (const float*)d_in[7];
  const float* wv_w = (const float*)d_in[8];
  const float* wv_b = (const float*)d_in[9];
  const float* wo_w = (const float*)d_in[10];
  const float* wo_b = (const float*)d_in[11];
  float* out = (float*)d_out;

  char* ws = (char*)d_ws;
  const size_t elems = (size_t)M * D;          // 4M elements
  bf16*  Qb = (bf16*)(ws);                     //  8 MB
  bf16*  Kb = (bf16*)(ws + elems * 2);         //  8 MB
  bf16*  Vb = (bf16*)(ws + elems * 4);         //  8 MB
  float* Cx = (float*)(ws + elems * 6);        // 16 MB

  dim3 blk(128);
  dim3 gg(M / 64, D / 64);        // 64 x 16 tiles

  gemm_bias_kernel<bf16><<<gg, blk, 0, stream>>>(q, wq_w, wq_b, Qb, M, D, D);
  gemm_bias_kernel<bf16><<<gg, blk, 0, stream>>>(k, wk_w, wk_b, Kb, M, D, D);
  gemm_bias_kernel<bf16><<<gg, blk, 0, stream>>>(v, wv_w, wv_b, Vb, M, D, D);

  dim3 ga(S / 64, Bn * 16);       // 32 q-blocks x 32 (b,h) pairs
  attn_kernel<<<ga, blk, 0, stream>>>(Qb, Kb, Vb, mask, Cx);

  gemm_bias_kernel<float><<<gg, blk, 0, stream>>>(Cx, wo_w, wo_b, out, M, D, D);
}
